// BoxPairHead_8177617732033
// MI455X (gfx1250) — compile-verified, matrix operand order checked by reading, not run on hardware
//
#include <hip/hip_runtime.h>
#include <math.h>

// ---------------------------------------------------------------------------
// BoxPairHead / bipartite GNN for MI455X (gfx1250, wave32, WMMA bf16)
//
// Dominant cost: adjacency layer-2 GEMM [131072,1024]@[1024,512] per iter.
// Layer-1 factored through [eu|ev]; layers 1-3 fused. Blocks of 128 pairs
// (4u x 32v), 16 waves; h1 staged in LDS in WMMA A-fragment order in two
// K-halves so the hot loop is ds_load_b128 + global_load_b128 + v_wmma.
// Small f32 GEMMs use LDS-tiled 32x32 blocking (L2 traffic /16).
// ---------------------------------------------------------------------------

typedef __bf16 bf16_t;
typedef __attribute__((ext_vector_type(16))) bf16_t v16bf;
typedef __attribute__((ext_vector_type(8)))  bf16_t v8bf;
typedef __attribute__((ext_vector_type(8)))  float  v8f;
typedef __attribute__((ext_vector_type(4)))  float  v4f;

#define NUc   256
#define NVc   512
#define KDIM  512
#define RDIM  1024
#define RH    512
#define NITER 2

#define NTIL  8                                   // A-tiles per block (4u x 2v)
#define TSTR  8192                                // elements per A-tile in LDS (16c*32l*16)
#define H1BYT ((size_t)NTIL * TSTR * 2)           // 128 KB
#define REDFL (16 * NTIL * 16)                    // 16 waves x 8 tiles x 16 m
#define SMEMB (H1BYT + REDFL * 4)                 // ~136 KB

__device__ __forceinline__ bf16_t f2bf(float f) {
  unsigned u = __float_as_uint(f);
  u += 0x7FFFu + ((u >> 16) & 1u);          // round-to-nearest-even
  unsigned short h = (unsigned short)(u >> 16);
  return __builtin_bit_cast(bf16_t, h);
}

__device__ __forceinline__ v16bf ldB(const bf16_t* __restrict__ p, int k1, int k2) {
  v8bf lo = *(const v8bf*)(p + k1);
  v8bf hi = *(const v8bf*)(p + k2);
  return __builtin_shufflevector(lo, hi, 0,1,2,3,4,5,6,7,8,9,10,11,12,13,14,15);
}

// ---------------------------------------------------------------------------
// W2 [1024,512] f32 row-major -> W2T [512,1024] bf16
// ---------------------------------------------------------------------------
__global__ __launch_bounds__(256)
void k_convert_w2t(const float* __restrict__ W2, bf16_t* __restrict__ W2T) {
  int idx = blockIdx.x * 256 + threadIdx.x;   // 512*1024 total
  int n = idx >> 10;
  int k = idx & 1023;
  W2T[idx] = f2bf(W2[(size_t)k * RH + n]);
}

// ---------------------------------------------------------------------------
// Fused adjacency kernel: 128 pairs (4u x 32v) per block, 512 thr = 16 waves.
//   h1 = relu(Au[u]+Av[v])  -> LDS (bf16, A-fragment order, K staged in halves)
//   h2 = relu(h1 @ W2 + b2) -> WMMA bf16, f32 accum
//   adj = sigmoid(h2 . W3 + b3)
// Wave w owns N in [32w, 32w+32): 2 B tiles x 8 A tiles = 16 accumulators.
// ---------------------------------------------------------------------------
__global__ __launch_bounds__(512)
void k_adj_fused(const float* __restrict__ Au, const float* __restrict__ Av,
                 const bf16_t* __restrict__ W2T, const float* __restrict__ b2,
                 const float* __restrict__ W3, const float* __restrict__ b3p,
                 float* __restrict__ adj) {
  extern __shared__ char smem_raw[];
  bf16_t* h1  = (bf16_t*)smem_raw;
  float*  red = (float*)(smem_raw + H1BYT);

  const int u0g  = blockIdx.y * 4;          // 4 u rows per block
  const int v0   = blockIdx.x * 32;         // 32 v cols per block
  const int tid  = threadIdx.x;
  const int lane = tid & 31;
  const int wave = tid >> 5;                // 0..15
  const int nl   = lane & 15;
  const int mh   = lane >> 4;
  const int kb   = mh << 3;

  const bf16_t* __restrict__ w0p = W2T + (size_t)(wave * 32 +  0 + nl) * RDIM;
  const bf16_t* __restrict__ w1p = W2T + (size_t)(wave * 32 + 16 + nl) * RDIM;

  v8f A0B0={},A0B1={},A1B0={},A1B1={},A2B0={},A2B1={},A3B0={},A3B1={};
  v8f A4B0={},A4B1={},A5B0={},A5B1={},A6B0={},A6B1={},A7B0={},A7B1={};

#define WM(dst, a, b) \
  dst = __builtin_amdgcn_wmma_f32_16x16x32_bf16(false, a, false, b, (short)0, dst, false, false)

  for (int kh = 0; kh < 2; ++kh) {
    const int kofs = kh * 512;

    // ---- fill h1 half-tile in LDS, A-fragment order ------------------------
    // slot s = (t, c, l): 16 bf16 (32B) = lane l's fragment piece of chunk c.
    for (int s = tid; s < NTIL * 16 * 32; s += 512) {
      int t = s >> 9;            // 0..7  (u = t>>1, v-tile = t&1)
      int c = (s >> 5) & 15;     // chunk within half
      int l = s & 31;
      const float* pu = Au + (size_t)(u0g + (t >> 1)) * RDIM;
      const float* pv = Av + (size_t)(v0 + (t & 1) * 16 + (l & 15)) * RDIM;
      int kb0 = kofs + c * 32 + ((l >> 4) << 3);
      v4f ua0 = *(const v4f*)(pu + kb0);
      v4f ua1 = *(const v4f*)(pu + kb0 + 4);
      v4f ua2 = *(const v4f*)(pu + kb0 + 16);
      v4f ua3 = *(const v4f*)(pu + kb0 + 20);
      v4f va0 = *(const v4f*)(pv + kb0);
      v4f va1 = *(const v4f*)(pv + kb0 + 4);
      v4f va2 = *(const v4f*)(pv + kb0 + 16);
      v4f va3 = *(const v4f*)(pv + kb0 + 20);
      v8bf lo, hi;
#pragma unroll
      for (int j = 0; j < 4; ++j) {
        lo[j]     = f2bf(fmaxf(ua0[j] + va0[j], 0.f));
        lo[4 + j] = f2bf(fmaxf(ua1[j] + va1[j], 0.f));
        hi[j]     = f2bf(fmaxf(ua2[j] + va2[j], 0.f));
        hi[4 + j] = f2bf(fmaxf(ua3[j] + va3[j], 0.f));
      }
      bf16_t* dst = h1 + ((size_t)t * TSTR + (size_t)c * 512 + (size_t)l * 16);
      *(v8bf*)dst       = lo;
      *(v8bf*)(dst + 8) = hi;
    }
    __syncthreads();

    // ---- WMMA over this K-half ---------------------------------------------
#pragma clang loop unroll_count(2)
    for (int c = 0; c < 16; ++c) {
      const int k1 = kofs + c * 32 + kb;
      const int k2 = k1 + 16;
      if (c + 1 < 16) {
        __builtin_prefetch((const void*)(w0p + k1 + 32), 0, 0);
        __builtin_prefetch((const void*)(w1p + k1 + 32), 0, 0);
      }
      v16bf b0 = ldB(w0p, k1, k2);
      v16bf b1 = ldB(w1p, k1, k2);
      const bf16_t* ap = h1 + (size_t)c * 512 + (size_t)lane * 16;
#define STEP(T)                                               \
      {                                                       \
        v16bf a_ = *(const v16bf*)(ap + (T) * TSTR);          \
        WM(A##T##B0, a_, b0);                                 \
        WM(A##T##B1, a_, b1);                                 \
      }
      STEP(0) STEP(1) STEP(2) STEP(3) STEP(4) STEP(5) STEP(6) STEP(7)
#undef STEP
    }
    __syncthreads();   // allow refill of h1 for next K-half
  }
#undef WM

  // ---- Epilogue: relu(+b2), dot W3, reduce over N, sigmoid -----------------
  const int n0 = wave * 32 + nl;
  const int n1 = n0 + 16;
  const float bb0 = b2[n0], wv0 = W3[n0];
  const float bb1 = b2[n1], wv1 = W3[n1];

#define EPI(T)                                                            \
  {                                                                       \
    float partial[8];                                                     \
    _Pragma("unroll") for (int i = 0; i < 8; ++i) {                       \
      float h0  = fmaxf(A##T##B0[i] + bb0, 0.f);                          \
      float h1v = fmaxf(A##T##B1[i] + bb1, 0.f);                          \
      partial[i] = h0 * wv0 + h1v * wv1;                                  \
    }                                                                     \
    _Pragma("unroll") for (int i = 0; i < 8; ++i) {                       \
      float v = partial[i];                                               \
      v += __shfl_xor(v, 1, 32);                                          \
      v += __shfl_xor(v, 2, 32);                                          \
      v += __shfl_xor(v, 4, 32);                                          \
      v += __shfl_xor(v, 8, 32);                                          \
      partial[i] = v;                                                     \
    }                                                                     \
    if (nl == 0) {                                                        \
      _Pragma("unroll") for (int i = 0; i < 8; ++i)                       \
          red[(wave * NTIL + (T)) * 16 + mh * 8 + i] = partial[i];        \
    }                                                                     \
  }
  EPI(0) EPI(1) EPI(2) EPI(3) EPI(4) EPI(5) EPI(6) EPI(7)
#undef EPI

  __syncthreads();
  if (tid < NTIL * 16) {
    int t = tid >> 4, m = tid & 15;
    float s = 0.f;
#pragma unroll
    for (int w = 0; w < 16; ++w) s += red[(w * NTIL + t) * 16 + m];
    s += b3p[0];
    int u = u0g + (t >> 1);
    int v = v0 + (t & 1) * 16 + m;
    adj[(size_t)u * NVc + v] = 1.f / (1.f + expf(-s));
  }
}

// ---------------------------------------------------------------------------
// LDS-tiled f32 GEMM: C = act( A1@B[0:K1] (+ A2@B[K1:K1+K2]) + bias )
// 32x32 output tile per block (16x16 threads, 2x2 per thread).
// transA applies to A1 only. M, N and all K are multiples of 32.
// ---------------------------------------------------------------------------
#define TS 32
__global__ __launch_bounds__(256)
void k_gemm_f32(const float* __restrict__ A1, int lda1, int K1, int transA,
                const float* __restrict__ A2, int lda2, int K2,
                const float* __restrict__ B, int ldb,
                const float* __restrict__ bias,
                float* __restrict__ C, int ldc, int relu) {
  __shared__ float As[TS][TS + 1];
  __shared__ float Bs[TS][TS + 1];
  const int tx = threadIdx.x, ty = threadIdx.y;
  const int m0 = blockIdx.y * TS, n0 = blockIdx.x * TS;
  const int eb = ty * 16 + tx;
  float acc00 = 0.f, acc01 = 0.f, acc10 = 0.f, acc11 = 0.f;

  for (int s = 0; s < 2; ++s) {
    const float* A = (s == 0) ? A1 : A2;
    if (!A) continue;                       // uniform across the block
    const int K   = (s == 0) ? K1 : K2;
    const int lda = (s == 0) ? lda1 : lda2;
    const int kof = (s == 0) ? 0 : K1;
    const int tA  = (s == 0) ? transA : 0;
    for (int kt = 0; kt < K; kt += TS) {
#pragma unroll
      for (int j = 0; j < 4; ++j) {
        int e = eb + j * 256;
        int r = e >> 5, cc = e & 31;
        As[r][cc] = tA ? A[(size_t)(kt + cc) * lda + (m0 + r)]
                       : A[(size_t)(m0 + r) * lda + (kt + cc)];
        Bs[r][cc] = B[(size_t)(kof + kt + r) * ldb + (n0 + cc)];
      }
      __syncthreads();
#pragma unroll
      for (int kk = 0; kk < TS; ++kk) {
        float a0 = As[ty][kk], a1 = As[ty + 16][kk];
        float b0 = Bs[kk][tx], b1 = Bs[kk][tx + 16];
        acc00 = fmaf(a0, b0, acc00);
        acc01 = fmaf(a0, b1, acc01);
        acc10 = fmaf(a1, b0, acc10);
        acc11 = fmaf(a1, b1, acc11);
      }
      __syncthreads();
    }
  }

  const float bi0 = bias ? bias[n0 + tx] : 0.f;
  const float bi1 = bias ? bias[n0 + tx + 16] : 0.f;
  float r00 = acc00 + bi0, r01 = acc01 + bi1;
  float r10 = acc10 + bi0, r11 = acc11 + bi1;
  if (relu) {
    r00 = fmaxf(r00, 0.f); r01 = fmaxf(r01, 0.f);
    r10 = fmaxf(r10, 0.f); r11 = fmaxf(r11, 0.f);
  }
  C[(size_t)(m0 + ty) * ldc + (n0 + tx)]           = r00;
  C[(size_t)(m0 + ty) * ldc + (n0 + tx + 16)]      = r01;
  C[(size_t)(m0 + ty + 16) * ldc + (n0 + tx)]      = r10;
  C[(size_t)(m0 + ty + 16) * ldc + (n0 + tx + 16)] = r11;
}

// ---------------------------------------------------------------------------
// Row LayerNorm over width W, 1 block per row. In-place safe.
// ---------------------------------------------------------------------------
__global__ __launch_bounds__(256)
void k_layernorm(const float* __restrict__ xin, float* __restrict__ xout,
                 const float* __restrict__ g, const float* __restrict__ b,
                 int W) {
  const int row = blockIdx.x;
  const float* p = xin + (size_t)row * W;
  float* q = xout + (size_t)row * W;
  float s = 0.f, s2 = 0.f;
  for (int i = threadIdx.x; i < W; i += 256) {
    float v = p[i];
    s += v;
    s2 = fmaf(v, v, s2);
  }
  __shared__ float sh1[256], sh2[256];
  sh1[threadIdx.x] = s;
  sh2[threadIdx.x] = s2;
  __syncthreads();
  for (int st = 128; st > 0; st >>= 1) {
    if (threadIdx.x < st) {
      sh1[threadIdx.x] += sh1[threadIdx.x + st];
      sh2[threadIdx.x] += sh2[threadIdx.x + st];
    }
    __syncthreads();
  }
  float mean = sh1[0] / W;
  float var  = sh2[0] / W - mean * mean;
  float inv  = rsqrtf(var + 1e-5f);
  for (int i = threadIdx.x; i < W; i += 256)
    q[i] = (p[i] - mean) * inv * g[i] + b[i];
}

// ---------------------------------------------------------------------------
extern "C" void kernel_launch(void* const* d_in, const int* in_sizes, int n_in,
                              void* d_out, int out_size, void* d_ws,
                              size_t ws_size, hipStream_t stream) {
  const float* enc_u  = (const float*)d_in[0];
  const float* enc_v  = (const float*)d_in[1];
  const float* adjW1  = (const float*)d_in[2];
  const float* adjb1  = (const float*)d_in[3];
  const float* adjW2  = (const float*)d_in[4];
  const float* adjb2  = (const float*)d_in[5];
  const float* adjW3  = (const float*)d_in[6];
  const float* adjb3  = (const float*)d_in[7];
  const float* utovW  = (const float*)d_in[8];
  const float* utovb  = (const float*)d_in[9];
  const float* vtouW  = (const float*)d_in[10];
  const float* vtoub  = (const float*)d_in[11];
  const float* utov_g = (const float*)d_in[12];
  const float* utov_b = (const float*)d_in[13];
  const float* vtou_g = (const float*)d_in[14];
  const float* vtou_b = (const float*)d_in[15];
  const float* uupdW  = (const float*)d_in[16];
  const float* uupd_g = (const float*)d_in[17];
  const float* uupd_b = (const float*)d_in[18];
  const float* vupdW  = (const float*)d_in[19];
  const float* vupd_g = (const float*)d_in[20];
  const float* vupd_b = (const float*)d_in[21];

  // ---- workspace carve-out -------------------------------------------------
  char* ws = (char*)d_ws;
  size_t off = 0;
  auto carve = [&](size_t bytes) {
    size_t r = off;
    off = (off + bytes + 255) & ~(size_t)255;
    return r;
  };
  float*  Au  = (float*)(ws + carve((size_t)NUc * RDIM * 4));
  float*  Av  = (float*)(ws + carve((size_t)NVc * RDIM * 4));
  bf16_t* W2T = (bf16_t*)(ws + carve((size_t)RH * RDIM * 2));
  float*  eu  = (float*)(ws + carve((size_t)NUc * KDIM * 4));
  float*  ev  = (float*)(ws + carve((size_t)NVc * KDIM * 4));
  float*  Mv  = (float*)(ws + carve((size_t)NVc * RDIM * 4));
  float*  Mu  = (float*)(ws + carve((size_t)NUc * RDIM * 4));
  float*  T1  = (float*)(ws + carve((size_t)NVc * RDIM * 4));
  float*  T2  = (float*)(ws + carve((size_t)NVc * KDIM * 4));

  float* outP = (float*)d_out;
  float* adjP = outP + (size_t)NUc * KDIM + (size_t)NVc * KDIM;  // adj slot

  auto gemm = [&](const float* A1, int lda1, int K1, int transA,
                  const float* A2, int lda2, int K2, const float* B, int ldb,
                  const float* bias, float* C, int M, int N, int relu) {
    k_gemm_f32<<<dim3(N / TS, M / TS), dim3(16, 16), 0, stream>>>(
        A1, lda1, K1, transA, A2, lda2, K2, B, ldb, bias, C, N, relu);
  };

  // working copies of encodings (inputs must stay pristine)
  hipMemcpyAsync(eu, enc_u, (size_t)NUc * KDIM * 4, hipMemcpyDeviceToDevice, stream);
  hipMemcpyAsync(ev, enc_v, (size_t)NVc * KDIM * 4, hipMemcpyDeviceToDevice, stream);
  k_convert_w2t<<<(RH * RDIM) / 256, 256, 0, stream>>>(adjW2, W2T);

  for (int it = 0; it < NITER; ++it) {
    // Au = eu @ W1[:K] + b1 ; Av = ev @ W1[K:]   (layer-1 factorization)
    gemm(eu, KDIM, KDIM, 0, nullptr, 0, 0, adjW1, RDIM, adjb1, Au, NUc, RDIM, 0);
    gemm(ev, KDIM, KDIM, 0, nullptr, 0, 0, adjW1 + (size_t)KDIM * RDIM, RDIM,
         nullptr, Av, NVc, RDIM, 0);
    // fused adjacency MLP (WMMA bf16) -> adj
    k_adj_fused<<<dim3(NVc / 32, NUc / 4), 512, SMEMB, stream>>>(
        Au, Av, W2T, adjb2, adjW3, adjb3, adjP);
    // v -> u message, u update
    gemm(ev, KDIM, KDIM, 0, nullptr, 0, 0, vtouW, RDIM, vtoub, Mv, NVc, RDIM, 1);
    gemm(adjP, NVc, NVc, 0, nullptr, 0, 0, Mv, RDIM, nullptr, T1, NUc, RDIM, 0);
    k_layernorm<<<NUc, 256, 0, stream>>>(T1, T1, vtou_g, vtou_b, RDIM);
    gemm(eu, KDIM, KDIM, 0, T1, RDIM, RDIM, uupdW, KDIM, nullptr, T2, NUc, KDIM, 0);
    k_layernorm<<<NUc, 256, 0, stream>>>(T2, eu, uupd_g, uupd_b, KDIM);
    // u -> v message (uses UPDATED eu), v update
    gemm(eu, KDIM, KDIM, 0, nullptr, 0, 0, utovW, RDIM, utovb, Mu, NUc, RDIM, 1);
    gemm(adjP, NVc, NUc, 1, nullptr, 0, 0, Mu, RDIM, nullptr, T1, NVc, RDIM, 0);
    k_layernorm<<<NVc, 256, 0, stream>>>(T1, T1, utov_g, utov_b, RDIM);
    gemm(ev, KDIM, KDIM, 0, T1, RDIM, RDIM, vupdW, KDIM, nullptr, T2, NVc, KDIM, 0);
    k_layernorm<<<NVc, 256, 0, stream>>>(T2, ev, vupd_g, vupd_b, KDIM);
  }

  // outputs: [eu | ev | adj] (adj already written in place)
  hipMemcpyAsync(outP, eu, (size_t)NUc * KDIM * 4, hipMemcpyDeviceToDevice, stream);
  hipMemcpyAsync(outP + (size_t)NUc * KDIM, ev, (size_t)NVc * KDIM * 4,
                 hipMemcpyDeviceToDevice, stream);
}
